// mAP_18167711662362
// MI455X (gfx1250) — matrix operand order repaired and checked by reading, not compiled, not run on hardware
//
#include <hip/hip_runtime.h>

#define PN 8192
#define GN 4096
#define NCLS 3            // classes 1..3
#define EPSV 1e-5f
#define IOU_THR 0.5f

typedef __attribute__((ext_vector_type(16))) _Float16 v16h;
typedef __attribute__((ext_vector_type(8)))  float    v8f;

// ---------------- workspace layout (bytes) ----------------
// int      order[NCLS*PN]
// _Float16 tpb[NCLS*PN]
// _Float16 fpb[NCLS*PN]
// int      npred[NCLS]
// int      ngt[NCLS]
// float    ap[NCLS]
#define WS_ORDER 0
#define WS_TPB   (WS_ORDER + NCLS*PN*4)
#define WS_FPB   (WS_TPB   + NCLS*PN*2)
#define WS_NPRED (WS_FPB   + NCLS*PN*2)
#define WS_NGT   (WS_NPRED + NCLS*4)
#define WS_AP    (WS_NGT   + NCLS*4)

// ---------------- kernel 1: per-class counts ----------------
__global__ __launch_bounds__(1024) void stats_kernel(const int* __restrict__ labels,
                                                     const float* __restrict__ gt,
                                                     int* __restrict__ npred,
                                                     int* __restrict__ ngt) {
    __shared__ int cnt[6];
    int tid = threadIdx.x;
    if (tid < 6) cnt[tid] = 0;
    __syncthreads();
    for (int i = tid; i < PN; i += 1024) {
        int l = labels[i];
        if (l >= 1 && l <= NCLS) atomicAdd(&cnt[l - 1], 1);
    }
    for (int g = tid; g < GN; g += 1024) {
        float lf = gt[g * 5];
        int l = (int)lf;
        if (l >= 1 && l <= NCLS && lf == (float)l) atomicAdd(&cnt[3 + l - 1], 1);
    }
    __syncthreads();
    if (tid < NCLS) npred[tid] = cnt[tid];
    if (tid >= 3 && tid < 3 + NCLS) ngt[tid - 3] = cnt[tid];
}

// ---------------- kernel 2: per-class bitonic sort (descending) ----------------
__global__ __launch_bounds__(1024) void sort_kernel(const float* __restrict__ probits,
                                                    const int* __restrict__ labels,
                                                    int* __restrict__ order) {
    __shared__ float skey[PN];
    __shared__ int   sidx[PN];
    int c   = blockIdx.x + 1;
    int tid = threadIdx.x;
    for (int i = tid; i < PN; i += 1024) {
        skey[i] = (labels[i] == c) ? probits[i] : -__builtin_inff();
        sidx[i] = i;
    }
    __syncthreads();
    for (int k = 2; k <= PN; k <<= 1) {
        for (int j = k >> 1; j > 0; j >>= 1) {
            for (int i = tid; i < PN; i += 1024) {
                int ixj = i ^ j;
                if (ixj > i) {
                    bool up = ((i & k) == 0);
                    if ((skey[i] < skey[ixj]) == up) {   // descending order
                        float tk = skey[i]; skey[i] = skey[ixj]; skey[ixj] = tk;
                        int   ti = sidx[i]; sidx[i] = sidx[ixj]; sidx[ixj] = ti;
                    }
                }
            }
            __syncthreads();
        }
    }
    for (int i = tid; i < PN; i += 1024) order[blockIdx.x * PN + i] = sidx[i];
}

// ---------------- kernel 3: sequential greedy-matching scan ----------------
__global__ __launch_bounds__(256) void scan_kernel(const float* __restrict__ pred_boxes,
                                                   const float* __restrict__ gt,
                                                   const int* __restrict__ order,
                                                   const int* __restrict__ npred,
                                                   _Float16* __restrict__ tpb,
                                                   _Float16* __restrict__ fpb) {
    __shared__ float sval[256];
    __shared__ int   sgi[256];
    __shared__ int   scnt[256];
    __shared__ unsigned char gmatch[GN];
    int cls = blockIdx.x;
    int c   = cls + 1;
    int tid = threadIdx.x;
    for (int g = tid; g < GN; g += 256) gmatch[g] = 0;

    // each thread owns 16 contiguous GT boxes in registers
    float gx0[16], gy0[16], gx1[16], gy1[16], ga[16];
    unsigned int gm = 0;
    int base = tid * 16;
#pragma unroll
    for (int u = 0; u < 16; ++u) {
        int g = base + u;
        float lf = gt[g * 5 + 0];
        gx0[u] = gt[g * 5 + 1]; gy0[u] = gt[g * 5 + 2];
        gx1[u] = gt[g * 5 + 3]; gy1[u] = gt[g * 5 + 4];
        ga[u]  = (gx1[u] - gx0[u]) * (gy1[u] - gy0[u]);
        if (lf == (float)c) gm |= (1u << u);
    }
    int npredc = npred[cls];
    const int* ord = order + cls * PN;
    __syncthreads();

    for (int j = 0; j < PN; ++j) {
        if (j >= npredc) {                      // uniform: all remaining invalid -> zero fill
            for (int jj = j + tid; jj < PN; jj += 256) {
                tpb[cls * PN + jj] = (_Float16)0.0f;
                fpb[cls * PN + jj] = (_Float16)0.0f;
            }
            break;
        }
        int p = ord[j];
        float px0 = pred_boxes[p * 4 + 0], py0 = pred_boxes[p * 4 + 1];
        float px1 = pred_boxes[p * 4 + 2], py1 = pred_boxes[p * 4 + 3];
        float pa  = (px1 - px0) * (py1 - py0);

        float bv = -2.0f; int bg = base; unsigned int cm = 0;
#pragma unroll
        for (int u = 0; u < 16; ++u) {
            float ltx = fmaxf(gx0[u], px0);
            float lty = fmaxf(gy0[u], py0);
            float rbx = fminf(gx1[u], px1);
            float rby = fminf(gy1[u], py1);
            float w = fmaxf(rbx - ltx, 0.0f);
            float h = fmaxf(rby - lty, 0.0f);
            float inter = w * h;
            float iou = inter / (ga[u] + pa - inter);
            bool m = (((gm >> u) & 1u) != 0u) && (iou > IOU_THR);
            float v = m ? iou : -1.0f;
            if (m) cm |= (1u << u);
            if (v > bv) { bv = v; bg = base + u; }   // strict > keeps first occurrence
        }
        sval[tid] = bv; sgi[tid] = bg;
        __syncthreads();
        for (int s = 128; s > 0; s >>= 1) {          // first-occurrence argmax reduce
            if (tid < s) {
                float v2 = sval[tid + s]; int g2 = sgi[tid + s];
                if (v2 > sval[tid] || (v2 == sval[tid] && g2 < sgi[tid])) {
                    sval[tid] = v2; sgi[tid] = g2;
                }
            }
            __syncthreads();
        }
        int   gidx = sgi[0];
        float maxv = sval[0];
        int cnt = 0;                                  // rank = popcount(mask below gidx)
        if (base + 16 <= gidx)      cnt = __popc(cm);
        else if (base < gidx)       cnt = __popc(cm & ((1u << (gidx - base)) - 1u));
        scnt[tid] = cnt;
        __syncthreads();
        for (int s = 128; s > 0; s >>= 1) {
            if (tid < s) scnt[tid] += scnt[tid + s];
            __syncthreads();
        }
        if (tid == 0) {
            int rank = scnt[0];
            bool hit     = maxv > 0.0f;               // any(col_mask); valid==true here
            bool matched = gmatch[rank] != 0;
            bool tp = hit && !matched;
            bool fp = !tp;                            // valid && !tp
            if (tp) gmatch[rank] = 1;
            tpb[cls * PN + j] = (_Float16)(tp ? 1.0f : 0.0f);
            fpb[cls * PN + j] = (_Float16)(fp ? 1.0f : 0.0f);
        }
        __syncthreads();
    }
}

// ---------------- kernel 4: WMMA prefix sums + AP ----------------
__global__ __launch_bounds__(256) void ap_kernel(const _Float16* __restrict__ tpb,
                                                 const _Float16* __restrict__ fpb,
                                                 const int* __restrict__ npred,
                                                 const int* __restrict__ ngt,
                                                 float* __restrict__ ap) {
    __shared__ unsigned short prefT[PN];
    __shared__ unsigned short prefF[PN];
    __shared__ int   rowT[PN / 16];
    __shared__ int   rowF[PN / 16];
    __shared__ float red[256];
    int cls  = blockIdx.x;
    int tid  = threadIdx.x;
    int lane = tid & 31;
    int wave = tid >> 5;
    int lhalf = (lane >= 16) ? 1 : 0;
    int lcol  = lane & 15;

    // B = inclusive upper-triangular T (K x N, T[k][n] = k<=n), K rows 16..31 zero.
    // 16-bit B 32x16 layout: lanes 0-15 hold K=0..7 (halves 0..7) of column N=lane; lanes 16-31 hold K=8..15; halves 8..15 (K=16..31) zero.
    v16h bmat = {};
    {
#pragma unroll
        for (int h = 0; h < 8; ++h)
            bmat[h] = (lhalf * 8 + h <= lcol) ? (_Float16)1.0f : (_Float16)0.0f;
    }

    // 64 tasks: 32 row-groups (of 16 rows x 16 elems) for tp, 32 for fp; 8 waves share.
    for (int t = wave; t < 64; t += 8) {
        int arr = t >> 5;
        int grp = t & 31;
        const _Float16* src = (arr == 0 ? tpb : fpb) + cls * PN + grp * 256;
        // A 16x32 f16 layout: lane<16 -> row=lane, K=0..7 in halves 0..7 (K=16..23 zero);
        //                     lane>=16 -> row=lane-16, K=8..15 in halves 0..7 (K=24..31 zero).
        v16h a = {};
        int off = lcol * 16 + lhalf * 8;
#pragma unroll
        for (int u = 0; u < 8; ++u) a[u] = src[off + u];
        v8f cacc = {};
        v8f d = __builtin_amdgcn_wmma_f32_16x16x32_f16(false, a, false, bmat,
                                                       (short)0, cacc, false, false);
        unsigned short* dstp = (arr == 0) ? prefT : prefF;
#pragma unroll
        for (int i = 0; i < 8; ++i) {            // D layout: lane<16 rows 0..7, lane>=16 rows 8..15
            int row = grp * 16 + lhalf * 8 + i;
            dstp[row * 16 + lcol] = (unsigned short)(d[i] + 0.5f);  // exact small integers
        }
    }
    __syncthreads();

    if (tid == 0) {                               // carry across 512 rows (exact ints)
        int run = 0;
        for (int r = 0; r < PN / 16; ++r) { rowT[r] = run; run += prefT[r * 16 + 15]; }
    }
    if (tid == 1) {
        int run = 0;
        for (int r = 0; r < PN / 16; ++r) { rowF[r] = run; run += prefF[r * 16 + 15]; }
    }
    __syncthreads();

    int   npredc = npred[cls];
    float ngtf   = (float)ngt[cls];
    float acc = 0.0f;
    for (int j = tid; j < PN; j += 256) {
        if (j < npredc) {
            float tpc = (float)((int)prefT[j] + rowT[j >> 4]);
            float fpc = (float)((int)prefF[j] + rowF[j >> 4]);
            float prec = tpc / (tpc + fpc + EPSV);
            float rec  = fpc / (ngtf + EPSV);     // reference uses fpc here — replicate
            acc += prec / rec;
        }
    }
    red[tid] = acc;
    __syncthreads();
    for (int s = 128; s > 0; s >>= 1) {
        if (tid < s) red[tid] += red[tid + s];
        __syncthreads();
    }
    if (tid == 0) {
        int np = npredc;
        float a2 = red[0] / (float)(np > 0 ? np : 1);
        ap[cls] = (np > 0 && ngt[cls] > 0) ? a2 : 0.0f;
    }
}

// ---------------- kernel 5: mean of APs ----------------
__global__ void final_kernel(const float* __restrict__ ap, float* __restrict__ out) {
    out[0] = (ap[0] + ap[1] + ap[2]) / 3.0f;
}

extern "C" void kernel_launch(void* const* d_in, const int* in_sizes, int n_in,
                              void* d_out, int out_size, void* d_ws, size_t ws_size,
                              hipStream_t stream) {
    (void)in_sizes; (void)n_in; (void)out_size; (void)ws_size;
    const float* probits = (const float*)d_in[0];
    const float* pboxes  = (const float*)d_in[1];
    const int*   plabels = (const int*)d_in[2];
    const float* gt      = (const float*)d_in[3];

    char* ws = (char*)d_ws;
    int*      order = (int*)(ws + WS_ORDER);
    _Float16* tpb   = (_Float16*)(ws + WS_TPB);
    _Float16* fpb   = (_Float16*)(ws + WS_FPB);
    int*      npred = (int*)(ws + WS_NPRED);
    int*      ngt   = (int*)(ws + WS_NGT);
    float*    ap    = (float*)(ws + WS_AP);

    stats_kernel<<<dim3(1), dim3(1024), 0, stream>>>(plabels, gt, npred, ngt);
    sort_kernel <<<dim3(NCLS), dim3(1024), 0, stream>>>(probits, plabels, order);
    scan_kernel <<<dim3(NCLS), dim3(256), 0, stream>>>(pboxes, gt, order, npred, tpb, fpb);
    ap_kernel   <<<dim3(NCLS), dim3(256), 0, stream>>>(tpb, fpb, npred, ngt, ap);
    final_kernel<<<dim3(1), dim3(1), 0, stream>>>(ap, (float*)d_out);
}
